// ArSSR_46291157517020
// MI455X (gfx1250) — matrix operand order repaired and checked by reading, not compiled, not run on hardware
//
#include <hip/hip_runtime.h>
#include <hip/hip_bf16.h>

// ---------------------------------------------------------------------------
// ArSSR fused pipeline for gfx1250 (MI455X):
//   prep_kernel : fp32 weights -> f16, transposed [N][K], padded
//   conv_kernel : 3^3 conv, 1->128 ch, channel-last f16 feature map
//   mlp_kernel  : trilinear gather + 8-layer MLP with v_wmma_f32_16x16x32_f16,
//                 weights staged via global_load_async_to_lds_b128 (ASYNCcnt)
// ---------------------------------------------------------------------------

typedef _Float16 v16h __attribute__((ext_vector_type(16)));
typedef _Float16 v8h  __attribute__((ext_vector_type(8)));
typedef float    v8f  __attribute__((ext_vector_type(8)));

#define NBATCH   2
#define KPTS     262144          // 64^3
#define MROWS    (NBATCH*KPTS)   // 524288
#define HL       32
#define FEAT     128
#define M_TILE   128             // rows per workgroup (8 waves x 16)

#define USE_ASYNC_COPY 1         // global_load_async_to_lds_b128 weight staging

// workspace layout (bytes)
#define FM_HALFS   (NBATCH*HL*HL*HL*FEAT)      // 8388608 halfs = 16 MB
#define WT_OFF_B   (FM_HALFS*2)                // 16777216
#define WT_HALFS   389120
#define BIAS_OFF_B (WT_OFF_B + WT_HALFS*2)     // + 778240

// per-layer weight offsets (halfs) inside the wt blob
__device__ __host__ constexpr int wt_off(int l) {
    constexpr int OFF[8] = {0, 40960, 106496, 172032, 212992, 253952, 319488, 385024};
    return OFF[l];
}

// ---------------------------------------------------------------------------
// prep: transpose+pad weights to f16 [Npad][Kpad]; pad biases to 256 f32/layer
// ---------------------------------------------------------------------------
struct PrepArgs {
    const float* w[8];
    const float* b[8];
};

__global__ __launch_bounds__(256) void prep_kernel(PrepArgs args,
                                                   _Float16* __restrict__ wt,
                                                   float* __restrict__ bias) {
    const int KPAD[8] = {160, 256, 256, 256, 160, 256, 256, 256};
    const int NPAD[8] = {256, 256, 256, 160, 256, 256, 256, 16};
    const int KACT[8] = {131, 256, 256, 256, 131, 256, 256, 256};
    const int NACT[8] = {256, 256, 256, 131, 256, 256, 256, 1};
    const int OFF[8]  = {0, 40960, 106496, 172032, 212992, 253952, 319488, 385024};

    int l = blockIdx.y;
    int kpad = KPAD[l], nact = NACT[l], kact = KACT[l];
    int sz = kpad * NPAD[l];
    const float* W = args.w[l];
    for (int i = blockIdx.x * blockDim.x + threadIdx.x; i < sz;
         i += gridDim.x * blockDim.x) {
        int nn = i / kpad;
        int kk = i - nn * kpad;
        float v = (kk < kact && nn < nact) ? W[kk * nact + nn] : 0.0f;
        wt[OFF[l] + i] = (_Float16)v;
    }
    if (blockIdx.x == 0) {
        const float* B = args.b[l];
        for (int j = threadIdx.x; j < 256; j += blockDim.x)
            bias[l * 256 + j] = (j < nact) ? B[j] : 0.0f;
    }
}

// ---------------------------------------------------------------------------
// conv: 3x3x3 SAME, 1 -> 128 channels, channel-last f16 output
// ---------------------------------------------------------------------------
__global__ __launch_bounds__(256) void conv_kernel(const float* __restrict__ img,
                                                   const float* __restrict__ cw,
                                                   const float* __restrict__ cb,
                                                   _Float16* __restrict__ fm) {
    int g = blockIdx.x * 256 + threadIdx.x;      // 8,388,608 threads total
    int c = g & 127;
    int rest = g >> 7;
    int x = rest & 31, y = (rest >> 5) & 31, z = (rest >> 10) & 31, n = rest >> 15;
    const float* im = img + n * (HL * HL * HL);
    const float* w  = cw + c * 27;
    float acc = cb[c];
#pragma unroll
    for (int kz = 0; kz < 3; ++kz)
#pragma unroll
        for (int ky = 0; ky < 3; ++ky)
#pragma unroll
            for (int kx = 0; kx < 3; ++kx) {
                int zz = z + kz - 1, yy = y + ky - 1, xx = x + kx - 1;
                if ((unsigned)zz < 32u && (unsigned)yy < 32u && (unsigned)xx < 32u)
                    acc += im[(zz * 32 + yy) * 32 + xx] * w[(kz * 3 + ky) * 3 + kx];
            }
    fm[g] = (_Float16)acc;   // index == (((n*32+z)*32+y)*32+x)*128 + c
}

// ---------------------------------------------------------------------------
// WMMA layer:  D(16x16) = A(16x32) * B(32x16) accumulated over KTILES
// A: lane = M row (lane&15), hi=lane>>4 picks k 8*hi+[0,8) and 16+8*hi+[0,8) -> 2x ds_load_b128
// B: weights stored [N][K] -> same two 8-half runs per lane as A
// C/D: element r -> row (hi*8+r), col (lane&15)
// ---------------------------------------------------------------------------
template <int KTILES, int NTILES, bool FINAL>
__device__ __forceinline__ void layer_gemm(const _Float16* __restrict__ s_in, int astr,
                                           const _Float16* __restrict__ sw, int kstr,
                                           _Float16* __restrict__ s_out, int ostr,
                                           const float* __restrict__ bias,
                                           float* __restrict__ gout, int growbase,
                                           int wave, int lane) {
    const int mbase = wave * 16;
    const int mrow  = lane & 15;
    const int hi    = lane >> 4;

    // cache this wave's A fragments for all K tiles in VGPRs
    v16h af[KTILES];
#pragma unroll
    for (int kt = 0; kt < KTILES; ++kt) {
        const _Float16* p = s_in + (mbase + mrow) * astr + kt * 32 + hi * 8;
        v8h lo = *(const v8h*)p;
        v8h hv = *(const v8h*)(p + 16);
        af[kt] = __builtin_shufflevector(lo, hv, 0, 1, 2, 3, 4, 5, 6, 7,
                                                 8, 9, 10, 11, 12, 13, 14, 15);
    }

    for (int nt = 0; nt < NTILES; ++nt) {
        v8f acc = {0.f, 0.f, 0.f, 0.f, 0.f, 0.f, 0.f, 0.f};
#pragma unroll
        for (int kt = 0; kt < KTILES; ++kt) {
            const _Float16* q = sw + (nt * 16 + mrow) * kstr + kt * 32 + hi * 8;
            v8h blo = *(const v8h*)q;
            v8h bhi = *(const v8h*)(q + 16);
            v16h bf = __builtin_shufflevector(blo, bhi, 0, 1, 2, 3, 4, 5, 6, 7,
                                                        8, 9, 10, 11, 12, 13, 14, 15);
            acc = __builtin_amdgcn_wmma_f32_16x16x32_f16(
                false, af[kt], false, bf, (short)0, acc, false, false);
        }
        if (FINAL) {
            if (mrow == 0) {            // only column 0 is the real output
                float b0 = bias[0];
#pragma unroll
                for (int r = 0; r < 8; ++r) {
                    float v = acc[r] + b0;
                    gout[growbase + mbase + hi * 8 + r] = v > 0.f ? v : 0.f;
                }
            }
        } else {
            float bn = bias[nt * 16 + mrow];
#pragma unroll
            for (int r = 0; r < 8; ++r) {
                float v = acc[r] + bn;
                v = v > 0.f ? v : 0.f;
                s_out[(mbase + hi * 8 + r) * ostr + nt * 16 + mrow] = (_Float16)v;
            }
        }
    }
}

// ---------------------------------------------------------------------------
// weight staging: global -> LDS
//   async path: GLOBAL_LOAD_ASYNC_TO_LDS_B128 (tracked by ASYNCcnt; direct
//   memory->LDS, no VGPR round-trip), completed with s_wait_asynccnt 0.
// ---------------------------------------------------------------------------
__device__ __forceinline__ void stage_weights(const _Float16* __restrict__ src,
                                              _Float16* __restrict__ dst,
                                              int halfs, int tid) {
    int n16 = halfs >> 3;                 // 16-byte chunks
#if USE_ASYNC_COPY
    for (int i = tid; i < n16; i += 256) {
        unsigned lds  = (unsigned)(uintptr_t)(dst + (i << 3)); // LDS byte addr (low 32b)
        unsigned goff = (unsigned)(i << 4);                    // byte offset from base
        asm volatile("global_load_async_to_lds_b128 %0, %1, %2 offset:0"
                     :: "v"(lds), "v"(goff), "s"(src)
                     : "memory");
    }
#else
    const uint4* s = (const uint4*)src;
    uint4* d = (uint4*)dst;
    for (int i = tid; i < n16; i += 256) d[i] = s[i];
#endif
}

__device__ __forceinline__ void wait_stage() {
#if USE_ASYNC_COPY
    asm volatile("s_wait_asynccnt 0x0" ::: "memory");
#endif
}

// prefetch next layer's weights into cache while current GEMM runs
__device__ __forceinline__ void prefetch_weights(const _Float16* __restrict__ src,
                                                 int halfs, int tid) {
    const char* p = (const char*)src;
    int bytes = halfs * 2;
    for (int off = tid * 128; off < bytes; off += 256 * 128)
        __builtin_prefetch(p + off, 0, 1);
}

// ---------------------------------------------------------------------------
// fused gather + MLP kernel: 8 waves, 128 rows per block
// ---------------------------------------------------------------------------
__global__ __launch_bounds__(256, 1) void mlp_kernel(const float* __restrict__ xyz,
                                                     const _Float16* __restrict__ fm,
                                                     const _Float16* __restrict__ wt,
                                                     const float* __restrict__ bias,
                                                     float* __restrict__ out) {
    __shared__ _Float16 s_x[M_TILE * 160];   // input x (feat+xyz+pad), then x+h
    __shared__ _Float16 s_a[M_TILE * 256];   // ping
    __shared__ _Float16 s_b[M_TILE * 256];   // pong
    __shared__ _Float16 s_w[256 * 256];      // current layer weights [N][K]

    const int tid  = threadIdx.x;
    const int lane = tid & 31;
    const int wave = tid >> 5;
    const int rowbase = blockIdx.x * M_TILE;

    // kick off layer-0 weight staging before the gather so the async DMA
    // overlaps the trilinear sampling work
    stage_weights(wt + wt_off(0), s_w, 40960, tid);

    // -------- phase 0: trilinear gather + concat into s_x --------
    {
        int r  = tid >> 1;        // 128 rows, 2 threads per row
        int cb = tid & 1;         // channel half: [cb*64, cb*64+64)
        int g  = rowbase + r;
        int n  = g >> 18;         // / 262144
        float gx = xyz[g * 3 + 0], gy = xyz[g * 3 + 1], gz = xyz[g * 3 + 2];
        // coords reversed: W-axis <- xyz[2], H-axis <- xyz[1], D-axis <- xyz[0]
        float sx = ((gz + 1.f) * 32.f - 1.f) * 0.5f;
        float sy = ((gy + 1.f) * 32.f - 1.f) * 0.5f;
        float sz = ((gx + 1.f) * 32.f - 1.f) * 0.5f;
        float fx0 = floorf(sx), fy0 = floorf(sy), fz0 = floorf(sz);
        float wx = sx - fx0, wy = sy - fy0, wz = sz - fz0;
        int x0 = (int)fx0, y0 = (int)fy0, z0 = (int)fz0;

        const _Float16* fmn = fm + (size_t)n * (HL * HL * HL * FEAT);
        int   coff[8];
        float cw8[8];
        int ci = 0;
#pragma unroll
        for (int dz = 0; dz < 2; ++dz)
#pragma unroll
            for (int dy = 0; dy < 2; ++dy)
#pragma unroll
                for (int dx = 0; dx < 2; ++dx, ++ci) {
                    int zi = z0 + dz, yi = y0 + dy, xi = x0 + dx;
                    float w = (dz ? wz : 1.f - wz) * (dy ? wy : 1.f - wy) *
                              (dx ? wx : 1.f - wx);
                    bool v = (unsigned)zi < 32u && (unsigned)yi < 32u &&
                             (unsigned)xi < 32u;
                    int zc = min(max(zi, 0), 31), yc = min(max(yi, 0), 31),
                        xc = min(max(xi, 0), 31);
                    coff[ci] = ((zc * 32 + yc) * 32 + xc) * FEAT;
                    cw8[ci]  = v ? w : 0.f;
                }
        // 64 channels in chunks of 8
        for (int ch = 0; ch < 8; ++ch) {
            float a8[8] = {0, 0, 0, 0, 0, 0, 0, 0};
#pragma unroll
            for (int k = 0; k < 8; ++k) {
                v8h d = *(const v8h*)(fmn + coff[k] + cb * 64 + ch * 8);
                float w = cw8[k];
#pragma unroll
                for (int j = 0; j < 8; ++j) a8[j] += w * (float)d[j];
            }
#pragma unroll
            for (int j = 0; j < 8; ++j)
                s_x[r * 160 + cb * 64 + ch * 8 + j] = (_Float16)a8[j];
        }
        if (cb == 0) {
#pragma unroll
            for (int j = 0; j < 32; ++j)
                s_x[r * 160 + 128 + j] =
                    (j < 3) ? (_Float16)xyz[g * 3 + j] : (_Float16)0.f;
        }
    }
    wait_stage();
    __syncthreads();

    // -------- stage 1 --------
    prefetch_weights(wt + wt_off(1), 65536, tid);
    layer_gemm<5, 16, false>(s_x, 160, s_w, 160, s_a, 256, bias + 0, nullptr, 0, wave, lane);
    __syncthreads();
    stage_weights(wt + wt_off(1), s_w, 65536, tid);
    wait_stage();
    __syncthreads();
    prefetch_weights(wt + wt_off(2), 65536, tid);
    layer_gemm<8, 16, false>(s_a, 256, s_w, 256, s_b, 256, bias + 256, nullptr, 0, wave, lane);
    __syncthreads();
    stage_weights(wt + wt_off(2), s_w, 65536, tid);
    wait_stage();
    __syncthreads();
    prefetch_weights(wt + wt_off(3), 40960, tid);
    layer_gemm<8, 16, false>(s_b, 256, s_w, 256, s_a, 256, bias + 512, nullptr, 0, wave, lane);
    __syncthreads();
    stage_weights(wt + wt_off(3), s_w, 40960, tid);
    wait_stage();
    __syncthreads();
    prefetch_weights(wt + wt_off(4), 40960, tid);
    layer_gemm<8, 10, false>(s_a, 256, s_w, 256, s_b, 256, bias + 768, nullptr, 0, wave, lane);
    __syncthreads();

    // -------- residual: x <- x + h --------
    stage_weights(wt + wt_off(4), s_w, 40960, tid);
    for (int i = tid; i < M_TILE * 160; i += 256) {
        int rr = i / 160, cc = i - rr * 160;
        s_x[i] = (_Float16)((float)s_x[i] + (float)s_b[rr * 256 + cc]);
    }
    wait_stage();
    __syncthreads();

    // -------- stage 2 --------
    prefetch_weights(wt + wt_off(5), 65536, tid);
    layer_gemm<5, 16, false>(s_x, 160, s_w, 160, s_a, 256, bias + 1024, nullptr, 0, wave, lane);
    __syncthreads();
    stage_weights(wt + wt_off(5), s_w, 65536, tid);
    wait_stage();
    __syncthreads();
    prefetch_weights(wt + wt_off(6), 65536, tid);
    layer_gemm<8, 16, false>(s_a, 256, s_w, 256, s_b, 256, bias + 1280, nullptr, 0, wave, lane);
    __syncthreads();
    stage_weights(wt + wt_off(6), s_w, 65536, tid);
    wait_stage();
    __syncthreads();
    prefetch_weights(wt + wt_off(7), 4096, tid);
    layer_gemm<8, 16, false>(s_b, 256, s_w, 256, s_a, 256, bias + 1536, nullptr, 0, wave, lane);
    __syncthreads();
    stage_weights(wt + wt_off(7), s_w, 4096, tid);
    wait_stage();
    __syncthreads();
    layer_gemm<8, 1, true>(s_a, 256, s_w, 256, nullptr, 0, bias + 1792, out, rowbase, wave, lane);
}

// ---------------------------------------------------------------------------
extern "C" void kernel_launch(void* const* d_in, const int* in_sizes, int n_in,
                              void* d_out, int out_size, void* d_ws, size_t ws_size,
                              hipStream_t stream) {
    const float* img    = (const float*)d_in[0];
    const float* xyz    = (const float*)d_in[1];
    const float* conv_w = (const float*)d_in[2];
    const float* conv_b = (const float*)d_in[3];

    PrepArgs pa;
    for (int i = 0; i < 8; ++i) {
        pa.w[i] = (const float*)d_in[4 + 2 * i];
        pa.b[i] = (const float*)d_in[5 + 2 * i];
    }

    _Float16* fm   = (_Float16*)d_ws;
    _Float16* wt   = (_Float16*)((char*)d_ws + WT_OFF_B);
    float*    bias = (float*)((char*)d_ws + BIAS_OFF_B);

    prep_kernel<<<dim3(32, 8), 256, 0, stream>>>(pa, wt, bias);
    conv_kernel<<<(NBATCH * HL * HL * HL * FEAT) / 256, 256, 0, stream>>>(
        img, conv_w, conv_b, fm);
    mlp_kernel<<<MROWS / M_TILE, 256, 0, stream>>>(xyz, fm, wt, bias, (float*)d_out);
}
